// SessionLSTM_30709016166696
// MI455X (gfx1250) — compile-verified, matrix operand order checked by reading, not run on hardware
//
#include <hip/hip_runtime.h>
#include <stdint.h>

// ---------- types ----------
typedef __attribute__((ext_vector_type(16))) __bf16 v16bf;
typedef __attribute__((ext_vector_type(8)))  __bf16 v8bf;
typedef __attribute__((ext_vector_type(8)))  float  v8f;

#define DI __device__ __forceinline__

// fp32 <-> bf16 (round to nearest even), bit-level so we avoid header variance
DI unsigned short f2bf(float f) {
    union { float f; uint32_t u; } x; x.f = f;
    uint32_t u = x.u;
    uint32_t r = (u + 0x7FFFu + ((u >> 16) & 1u)) >> 16;
    return (unsigned short)r;
}
DI float bf2f(unsigned short b) {
    union { uint32_t u; float f; } x; x.u = ((uint32_t)b) << 16;
    return x.f;
}
DI float sigm(float x) { return 1.0f / (1.0f + __expf(-x)); }

// CDNA5 async Global->LDS copy (ASYNCcnt path). 16 bytes per lane, data
// bypasses VGPRs entirely. LDS generic addresses carry the byte offset in
// bits [31:0] (ISA 10.2), so truncating the generic pointer gives the
// wave-relative LDS address the instruction wants.
DI void async_copy_b128(void* lds_dst, const void* gsrc) {
    uint32_t ldsa = (uint32_t)(uintptr_t)lds_dst;
    asm volatile("global_load_async_to_lds_b128 %0, %1, off"
                 :: "v"(ldsa), "v"(gsrc) : "memory");
}
DI void wait_async0() { asm volatile("s_wait_asynccnt 0" ::: "memory"); }

// Model constants
static const int Bsz = 32, Tlen = 200, Edim = 256, Hdim = 1024, Vsz = 20000;

// ------------------------------------------------------------------
// Pack fp32 weight W[N][K] (row-major) into bf16 WMMA B-fragment tiles.
// Tile (kt, nt) covers k in [kt*32, kt*32+32), n in [nt*16, nt*16+16).
// Layout inside a tile: 512 halfs ordered [lane][16 halfs], where
//   lane L holds column n = nt*16 + (L&15),
//   half j -> k = kt*32 + ( j<8 ? 8*(L>>4)+j : 16 + 8*(L>>4) + (j-8) )
// so the kernel-side B fragment is ONE contiguous 32-byte load per lane.
// ------------------------------------------------------------------
__global__ void pack_b_kernel(const float* __restrict__ W, unsigned short* __restrict__ out,
                              int N, int K, int Ntiles, long total) {
    long idx = (long)blockIdx.x * blockDim.x + threadIdx.x;
    if (idx >= total) return;
    int  within = (int)(idx & 511);
    long tile   = idx >> 9;
    int  lane = within >> 4, j = within & 15;
    int  kt = (int)(tile / Ntiles), nt = (int)(tile % Ntiles);
    int  n  = nt * 16 + (lane & 15);
    int  kk = (j < 8) ? (8 * (lane >> 4) + j) : (16 + 8 * (lane >> 4) + (j - 8));
    int  k  = kt * 32 + kk;
    float v = (n < N && k < K) ? W[(long)n * K + k] : 0.0f;
    out[idx] = f2bf(v);
}

// x_bf16[(b*T+t)*E + e] = bf16(emb[ids[b*T+t]][e])
__global__ void gather_embed_kernel(const int* __restrict__ ids, const float* __restrict__ emb,
                                    unsigned short* __restrict__ x, long total) {
    long idx = (long)blockIdx.x * blockDim.x + threadIdx.x;
    if (idx >= total) return;
    long bt = idx >> 8;          // E = 256
    int  e  = (int)(idx & 255);
    int  id = ids[bt];
    x[idx] = f2bf(emb[(long)id * Edim + e]);
}

__global__ void zero_u32_kernel(uint32_t* p, long n) {
    long i = (long)blockIdx.x * blockDim.x + threadIdx.x;
    long stride = (long)gridDim.x * blockDim.x;
    for (; i < n; i += stride) p[i] = 0u;
}

// ------------------------------------------------------------------
// Fused LSTM step for one (t, layer): gates = x@Wih^T + h@Whh^T (+bias),
// then cell update + length masking, all in one kernel.
// Grid: 16 blocks x 256 threads. Block b owns H-columns [b*64, b*64+64)
// of ALL FOUR gates (so the nonlinearity fuses). 8 waves:
//   wave w -> gate g = w>>1, M-tile m = w&1; each wave does 4 N-tiles.
// h is double-buffered across t (other blocks read full h rows); c is
// updated in place (columns are owned exclusively per block).
// ------------------------------------------------------------------
__global__ void __launch_bounds__(256)
lstm_step_kernel(const unsigned short* __restrict__ Ax, long ax_stride, int Kx,
                 const unsigned short* __restrict__ Bih,
                 const unsigned short* __restrict__ Bhh,
                 const unsigned short* __restrict__ hprev,
                 const float* __restrict__ b_ih, const float* __restrict__ b_hh,
                 float* __restrict__ c_state, unsigned short* __restrict__ h_out,
                 unsigned short* __restrict__ seq_out,
                 const int* __restrict__ lengths, int t) {
    const int tid  = threadIdx.x;
    const int lane = tid & 31, wave = tid >> 5;
    const int g = wave >> 1, mtile = wave & 1;
    const int n0 = blockIdx.x * 64;
    const int hi = lane >> 4, lo = lane & 15;

    v8f acc[4];
    for (int i = 0; i < 4; ++i)
        for (int q = 0; q < 8; ++q) acc[i][q] = 0.0f;

    // ---- input contribution: A = x rows (M=32, K=Kx) ----
    {
        const unsigned short* arow = Ax + (long)(mtile * 16 + lo) * ax_stride;
        const int kchunks = Kx >> 5;
        for (int kt = 0; kt < kchunks; ++kt) {
            v8bf a0 = *(const v8bf*)(arow + kt * 32 + 8 * hi);
            v8bf a1 = *(const v8bf*)(arow + kt * 32 + 16 + 8 * hi);
            v16bf a = __builtin_shufflevector(a0, a1, 0,1,2,3,4,5,6,7,8,9,10,11,12,13,14,15);
            for (int nt = 0; nt < 4; ++nt) {
                int ntg = g * 64 + blockIdx.x * 4 + nt;            // of 256 tiles over 4H
                v16bf b = *(const v16bf*)(Bih + (((long)kt * 256 + ntg) << 9) + lane * 16);
                acc[nt] = __builtin_amdgcn_wmma_f32_16x16x32_bf16(
                    false, a, false, b, (short)0, acc[nt], false, false);
            }
        }
    }
    // ---- recurrent contribution: A = h_prev rows (M=32, K=1024) ----
    {
        const unsigned short* arow = hprev + (long)(mtile * 16 + lo) * Hdim;
        for (int kt = 0; kt < 32; ++kt) {
            v8bf a0 = *(const v8bf*)(arow + kt * 32 + 8 * hi);
            v8bf a1 = *(const v8bf*)(arow + kt * 32 + 16 + 8 * hi);
            v16bf a = __builtin_shufflevector(a0, a1, 0,1,2,3,4,5,6,7,8,9,10,11,12,13,14,15);
            for (int nt = 0; nt < 4; ++nt) {
                int ntg = g * 64 + blockIdx.x * 4 + nt;
                v16bf b = *(const v16bf*)(Bhh + (((long)kt * 256 + ntg) << 9) + lane * 16);
                acc[nt] = __builtin_amdgcn_wmma_f32_16x16x32_bf16(
                    false, a, false, b, (short)0, acc[nt], false, false);
            }
        }
    }

    // ---- exchange gate blocks through LDS so one thread sees i,f,g,o ----
    __shared__ float gl[4][32][68];   // [gate][batch row][col], padded
    for (int nt = 0; nt < 4; ++nt)
        for (int r = 0; r < 8; ++r)
            gl[g][mtile * 16 + r + 8 * hi][nt * 16 + lo] = acc[nt][r];
    __syncthreads();

    // ---- pointwise cell update with length masking ----
    for (int e = tid; e < 32 * 64; e += 256) {
        int m = e >> 6, n = e & 63;
        int col = n0 + n;
        float gi = gl[0][m][n] + b_ih[col]            + b_hh[col];
        float gf = gl[1][m][n] + b_ih[Hdim + col]     + b_hh[Hdim + col];
        float gg = gl[2][m][n] + b_ih[2 * Hdim + col] + b_hh[2 * Hdim + col];
        float go = gl[3][m][n] + b_ih[3 * Hdim + col] + b_hh[3 * Hdim + col];
        long cidx = (long)m * Hdim + col;
        float co = c_state[cidx];
        float ho = bf2f(hprev[cidx]);
        float iv = sigm(gi), fv = sigm(gf), gv = tanhf(gg), ov = sigm(go);
        float cn = fv * co + iv * gv;
        float hn = ov * tanhf(cn);
        bool  msk = t < lengths[m];
        float cr = msk ? cn : co;
        float hr = msk ? hn : ho;
        c_state[cidx] = cr;
        h_out[cidx]   = f2bf(hr);
        if (seq_out) seq_out[((long)m * Tlen + t) * Hdim + col] = f2bf(hr);
    }
}

// ------------------------------------------------------------------
// Projection GEMM: logits(6400 x 20000) = Hseq(6400x1024 bf16) x W_out^T + b.
// Block = 256 threads (8 waves) computes a 128x128 C tile; wave w owns
// M-tile w and all 8 N-tiles (8 v8f accumulators). A chunk (128x32) and
// B panel (8 pre-packed tiles) are staged in LDS via the CDNA5 async
// Global->LDS path with двух-buffering: step kt+1's panels are issued
// asynchronously (ASYNCcnt) before the WMMAs of step kt, so L2->LDS
// traffic overlaps the 64 WMMAs per K-step.
// ------------------------------------------------------------------
__global__ void __launch_bounds__(256)
proj_kernel(const unsigned short* __restrict__ Aseq,
            const unsigned short* __restrict__ Bp,     // packed [32 kt][1250 nt][512]
            const float* __restrict__ bias, float* __restrict__ out) {
    __shared__ alignas(16) unsigned short Abuf[2][128][32];
    __shared__ alignas(32) unsigned short Bbuf[2][8][512];

    const int tid  = threadIdx.x;
    const int lane = tid & 31, wave = tid >> 5;
    const int hi = lane >> 4, lo = lane & 15;
    const int col0 = blockIdx.x * 128, row0 = blockIdx.y * 128;

    v8f acc[8];
    for (int i = 0; i < 8; ++i)
        for (int q = 0; q < 8; ++q) acc[i][q] = 0.0f;

    // async-stage K-step kt into LDS buffer `buf` (16B per lane per issue)
    auto stage = [&](int kt, int buf) {
        // A: 128 rows x 32 halfs (8 KB)
        for (int i = 0; i < 2; ++i) {
            int c = tid + i * 256;
            int row = c >> 2, off8 = (c & 3) * 8;
            async_copy_b128(&Abuf[buf][row][off8],
                            Aseq + (long)(row0 + row) * Hdim + kt * 32 + off8);
        }
        // B: 8 packed tiles (8 KB). OOB tiles are skipped; their garbage
        // accumulators are never stored (fp32 store is col-guarded).
        for (int i = 0; i < 2; ++i) {
            int c = tid + i * 256;
            int tile = c >> 6, off8 = (c & 63) * 8;
            int ntg = (col0 >> 4) + tile;
            if (ntg < Vsz / 16)
                async_copy_b128(&Bbuf[buf][tile][off8],
                                Bp + (((long)kt * (Vsz / 16) + ntg) << 9) + off8);
        }
    };

    stage(0, 0);
    for (int kt = 0; kt < 32; ++kt) {
        const int cur = kt & 1;
        wait_async0();          // my async copies into `cur` are complete
        __syncthreads();        // everyone's are, and prior reads are done
        if (kt + 1 < 32) stage(kt + 1, cur ^ 1);   // overlap with compute

        v8bf a0 = *(const v8bf*)&Abuf[cur][wave * 16 + lo][8 * hi];
        v8bf a1 = *(const v8bf*)&Abuf[cur][wave * 16 + lo][16 + 8 * hi];
        v16bf a = __builtin_shufflevector(a0, a1, 0,1,2,3,4,5,6,7,8,9,10,11,12,13,14,15);
        for (int nt = 0; nt < 8; ++nt) {
            v16bf b = *(const v16bf*)&Bbuf[cur][nt][lane * 16];
            acc[nt] = __builtin_amdgcn_wmma_f32_16x16x32_bf16(
                false, a, false, b, (short)0, acc[nt], false, false);
        }
    }

    // epilogue: bias + store fp32 logits
    for (int nt = 0; nt < 8; ++nt) {
        int col = col0 + nt * 16 + lo;
        if (col < Vsz) {
            float bb = bias[col];
            for (int r = 0; r < 8; ++r) {
                long rm = row0 + wave * 16 + r + 8 * hi;
                out[rm * (long)Vsz + col] = acc[nt][r] + bb;
            }
        }
    }
}

// ------------------------------------------------------------------
extern "C" void kernel_launch(void* const* d_in, const int* in_sizes, int n_in,
                              void* d_out, int out_size, void* d_ws, size_t ws_size,
                              hipStream_t stream) {
    (void)in_sizes; (void)n_in; (void)out_size; (void)ws_size;

    const int*   ids     = (const int*)d_in[0];
    const int*   lengths = (const int*)d_in[1];
    const float* emb     = (const float*)d_in[2];
    const float* W_ih_0  = (const float*)d_in[3];
    const float* W_hh_0  = (const float*)d_in[4];
    const float* b_ih_0  = (const float*)d_in[5];
    const float* b_hh_0  = (const float*)d_in[6];
    const float* W_ih_1  = (const float*)d_in[7];
    const float* W_hh_1  = (const float*)d_in[8];
    const float* b_ih_1  = (const float*)d_in[9];
    const float* b_hh_1  = (const float*)d_in[10];
    const float* W_out   = (const float*)d_in[11];
    const float* b_out   = (const float*)d_in[12];
    float* out = (float*)d_out;

    // ---- workspace layout (bf16 packed weights all fit in the 192MB L2) ----
    char* ws = (char*)d_ws;
    size_t off = 0;
    auto take = [&](size_t bytes) -> char* {
        char* p = ws + off;
        off = (off + bytes + 255) & ~(size_t)255;
        return p;
    };
    unsigned short* xg   = (unsigned short*)take((size_t)Bsz * Tlen * Edim * 2);      // 3.3 MB
    unsigned short* pih0 = (unsigned short*)take((size_t)8  * 256  * 512 * 2);        // 2 MB
    unsigned short* phh0 = (unsigned short*)take((size_t)32 * 256  * 512 * 2);        // 8 MB
    unsigned short* pih1 = (unsigned short*)take((size_t)32 * 256  * 512 * 2);        // 8 MB
    unsigned short* phh1 = (unsigned short*)take((size_t)32 * 256  * 512 * 2);        // 8 MB
    unsigned short* pout = (unsigned short*)take((size_t)32 * 1250 * 512 * 2);        // 39 MB
    unsigned short* hbuf = (unsigned short*)take((size_t)4 * Bsz * Hdim * 2);         // h0[2],h1[2]
    float*          cbuf = (float*)take((size_t)2 * Bsz * Hdim * 4);                  // c0,c1
    unsigned short* hseq = (unsigned short*)take((size_t)Bsz * Tlen * Hdim * 2);      // 13 MB

    // ---- deterministic per-call init of recurrent state ----
    zero_u32_kernel<<<256, 256, 0, stream>>>((uint32_t*)hbuf, (long)(4 * Bsz * Hdim * 2) / 4);
    zero_u32_kernel<<<256, 256, 0, stream>>>((uint32_t*)cbuf, (long)(2 * Bsz * Hdim));

    // ---- pack weights to bf16 WMMA-fragment tiles ----
    {
        long t0 = 8L  * 256  * 512;
        long t1 = 32L * 256  * 512;
        long t2 = 32L * 1250 * 512;
        pack_b_kernel<<<(t0 + 255) / 256, 256, 0, stream>>>(W_ih_0, pih0, 4 * Hdim, Edim, 256, t0);
        pack_b_kernel<<<(t1 + 255) / 256, 256, 0, stream>>>(W_hh_0, phh0, 4 * Hdim, Hdim, 256, t1);
        pack_b_kernel<<<(t1 + 255) / 256, 256, 0, stream>>>(W_ih_1, pih1, 4 * Hdim, Hdim, 256, t1);
        pack_b_kernel<<<(t1 + 255) / 256, 256, 0, stream>>>(W_hh_1, phh1, 4 * Hdim, Hdim, 256, t1);
        pack_b_kernel<<<(t2 + 255) / 256, 256, 0, stream>>>(W_out,  pout, Vsz,      Hdim, 1250, t2);
    }
    // ---- embedding gather -> bf16 ----
    {
        long tot = (long)Bsz * Tlen * Edim;
        gather_embed_kernel<<<(tot + 255) / 256, 256, 0, stream>>>(ids, emb, xg, tot);
    }

    // ---- sequential recurrence: 2 fused kernels per timestep ----
    for (int t = 0; t < Tlen; ++t) {
        int p = t & 1;
        unsigned short* h0p = hbuf + (size_t)p        * Bsz * Hdim;
        unsigned short* h0n = hbuf + (size_t)(p ^ 1)  * Bsz * Hdim;
        unsigned short* h1p = hbuf + (size_t)(2 + p)  * Bsz * Hdim;
        unsigned short* h1n = hbuf + (size_t)(2 + (p ^ 1)) * Bsz * Hdim;
        lstm_step_kernel<<<16, 256, 0, stream>>>(
            xg + (size_t)t * Edim, (long)Tlen * Edim, Edim,
            pih0, phh0, h0p, b_ih_0, b_hh_0,
            cbuf, h0n, (unsigned short*)nullptr, lengths, t);
        lstm_step_kernel<<<16, 256, 0, stream>>>(
            h0n, (long)Hdim, Hdim,
            pih1, phh1, h1p, b_ih_1, b_hh_1,
            cbuf + (size_t)Bsz * Hdim, h1n, hseq, lengths, t);
    }

    // ---- vocab projection (262 GFLOP, WMMA bf16) ----
    proj_kernel<<<dim3((Vsz + 127) / 128, (Bsz * Tlen) / 128), 256, 0, stream>>>(
        hseq, pout, b_out, out);
}
